// CapsNet_38680475468115
// MI455X (gfx1250) — matrix operand order repaired and checked by reading, not compiled, not run on hardware
//
#include <hip/hip_runtime.h>
#include <hip/hip_bf16.h>

// ---------------------------------------------------------------------------
// CapsNet forward for MI455X (gfx1250, wave32, WMMA + TDM).
// conv1/conv2 as implicit-GEMM f16 WMMA (f32 accumulate), decoder as dense
// f16 WMMA GEMMs. Weights pre-converted to zero-padded f16 so every B tile is
// a regular 2D tile staged by the Tensor Data Mover (tensor_load_to_lds with
// the LDS pad feature), tracked with TENSORcnt. Routing/softmax/squash VALU.
// ---------------------------------------------------------------------------

typedef _Float16 half_t;
typedef __attribute__((ext_vector_type(16))) _Float16 v16h;
typedef __attribute__((ext_vector_type(8)))  float    v8f;
typedef __attribute__((ext_vector_type(4)))  float    f32x4;

#define TILE_M 64
#define TILE_N 64
#define TILE_K 32
#define LDSK   40   // 32 halfs + 8 pad (80B row stride, keeps 16B alignment)

#if __has_builtin(__builtin_amdgcn_tensor_load_to_lds) && \
    __has_builtin(__builtin_amdgcn_s_wait_tensorcnt)
#define HAVE_TDM 1
typedef __attribute__((ext_vector_type(4))) unsigned int u32x4;
typedef __attribute__((ext_vector_type(8))) int          i32x8;
typedef __attribute__((ext_vector_type(4))) int          i32x4;

// Stage a 64-row x 32-half tile of a row-major f16 matrix (row stride
// `stride_e` elements) into LDS at byte offset `lds_off`, inserting 16B of
// padding after every 64B row so the LDS layout matches LDSK=40 halfs.
// dim0_rem/dim1_rem: remaining tensor extent from the tile origin (OOB rows
// read as zero -> handles the N=784 tail tile of the decoder).
__device__ __forceinline__ void tdm_load_tile_b(const half_t* gsrc,
                                                unsigned lds_off, int stride_e,
                                                int dim0_rem, int dim1_rem) {
  unsigned long long ga = (unsigned long long)(uintptr_t)gsrc;
  u32x4 g0;
  g0[0] = 1u;                                         // count=1, load descriptor
  g0[1] = lds_off;                                    // LDS byte address
  g0[2] = (unsigned)(ga & 0xffffffffu);               // global_addr[31:0]
  g0[3] = (unsigned)((ga >> 32) & 0x1ffffffu)         // global_addr[56:32]
          | (2u << 30);                               // type = 2 ("image")
  unsigned long long td0 = (unsigned long long)(unsigned)dim0_rem;
  unsigned long long td1 = (unsigned long long)(unsigned)dim1_rem;
  unsigned long long st0 = (unsigned long long)(unsigned)stride_e;
  i32x8 g1;
  g1[0] = (int)((1u << 16)      // data_size = 2 bytes
                | (1u << 20)    // pad_enable
                | (3u << 22)    // pad_interval: every 16 DWORDs (64B row)
                | (3u << 25));  // pad_amount: 4 DWORDs (16B)
  g1[1] = (int)((td0 & 0xffffu) << 16);               // tensor_dim0 lo16
  g1[2] = (int)(((td0 >> 16) & 0xffffu) | ((td1 & 0xffffu) << 16));
  g1[3] = (int)(((td1 >> 16) & 0xffffu) | (32u << 16));  // tile_dim0 = 32
  g1[4] = (int)64u;                                   // tile_dim1 = 64
  g1[5] = (int)(st0 & 0xffffffffu);                   // tensor_dim0_stride
  g1[6] = (int)((st0 >> 32) & 0xffffu);
  g1[7] = 0;
  i32x4 gz = {0, 0, 0, 0};
  i32x8 gz8 = {0, 0, 0, 0, 0, 0, 0, 0};
  // amdgpu-toolchain (clang-23) 6-arg form
  __builtin_amdgcn_tensor_load_to_lds(g0, g1, gz, gz, gz8, 0);
}
#else
#define HAVE_TDM 0
#endif

// --- WMMA fragment loaders (wave32 layouts per CDNA5 ISA 7.12.2) -----------
__device__ __forceinline__ v16h lds_frag_a(const half_t* t, int mSub, int lane) {
  const half_t* p = t + (mSub + (lane & 15)) * LDSK + ((lane < 16) ? 0 : 8);
  union { v16h v; f32x4 f[2]; } u;
  u.f[0] = *(const f32x4*)p;
  u.f[1] = *(const f32x4*)(p + 16);
  return u.v;
}
__device__ __forceinline__ v16h lds_frag_b(const half_t* t, int nSub, int lane) {
  const half_t* p = t + (nSub + (lane & 15)) * LDSK + ((lane < 16) ? 0 : 8);
  union { v16h v; f32x4 f[2]; } u;
  u.f[0] = *(const f32x4*)p;
  u.f[1] = *(const f32x4*)(p + 16);
  return u.v;
}
__device__ __forceinline__ v8f wmma_f16(v16h a, v16h b, v8f c) {
  return __builtin_amdgcn_wmma_f32_16x16x32_f16(false, a, false, b, (short)0, c,
                                                false, false);
}

// Stage one 64x32-half B tile: TDM if available, else vector copy.
__device__ __forceinline__ void stage_b_tile(const half_t* wgt, int n0, int k0,
                                             int strideK, int Nrows,
                                             half_t* Bs, int row, int kq) {
#if HAVE_TDM
  if (threadIdx.x == 0) {
    tdm_load_tile_b(wgt + (size_t)n0 * strideK + k0,
                    (unsigned)(uintptr_t)&Bs[0], strideK, strideK - k0,
                    Nrows - n0);
  }
  (void)row; (void)kq;
#else
  int n = n0 + row;
  f32x4* dst = (f32x4*)&Bs[row * LDSK + kq];
  if (n < Nrows) {
    *dst = *(const f32x4*)(wgt + (size_t)n * strideK + k0 + kq);
  } else {
    f32x4 z = {0.f, 0.f, 0.f, 0.f};
    *dst = z;
  }
#endif
}

__device__ __forceinline__ void wait_b_tile() {
#if HAVE_TDM
  __builtin_amdgcn_s_wait_tensorcnt(0);
#endif
}

// --- implicit-GEMM conv (9x9, VALID), weights pre-converted f16 [256][KPAD] -
template <int CIN, int HIN, int WIN, int HOUT, int WOUT, int STRIDE, int MTOT,
          bool RELU, bool IN_F16, bool OUT_F16>
__global__ __launch_bounds__(256) void conv_igemm_wmma_kernel(
    const void* __restrict__ xin, const half_t* __restrict__ wgt,
    const float* __restrict__ bias, void* __restrict__ yout) {
  constexpr int KTOT = CIN * 81;
  constexpr int KPAD = ((KTOT + TILE_K - 1) / TILE_K) * TILE_K;
  constexpr int OPIX = HOUT * WOUT;
  constexpr int OC   = 256;

  __shared__ alignas(16) half_t As[TILE_M * LDSK];
  __shared__ alignas(16) half_t Bs[TILE_N * LDSK];

  const int tid  = threadIdx.x;
  const int lane = tid & 31;
  const int wave = tid >> 5;
  const int wr = wave >> 2, wc = wave & 3;   // 2x4 wave grid over 64x64 tile
  const int m0 = blockIdx.x * TILE_M;
  const int n0 = blockIdx.y * TILE_N;

  const float*  xf = (const float*)xin;
  const half_t* xh = (const half_t*)xin;

  v8f acc0 = {}; v8f acc1 = {};

  const int row = tid >> 2;          // 64 rows, 4 threads/row
  const int kq  = (tid & 3) * 8;     // 8 contiguous k per thread

  const int mA   = m0 + row;         // MTOT % 64 == 0 -> always in range
  const int bimg = mA / OPIX;
  const int pos  = mA - bimg * OPIX;
  const int oy   = pos / WOUT;
  const int ox   = pos - oy * WOUT;

  for (int k0 = 0; k0 < KPAD; k0 += TILE_K) {
    stage_b_tile(wgt, n0, k0, KPAD, OC, Bs, row, kq);
    // stage A tile (im2col gather, f32/f16 -> f16)
    {
      half_t* dst = &As[row * LDSK + kq];
#pragma unroll
      for (int j = 0; j < 8; ++j) {
        int kk = k0 + kq + j;
        bool kok = (KPAD == KTOT) ? true : (kk < KTOT);  // folds for conv2
        float v = 0.f;
        if (kok) {
          int ic = kk / 81; int r2 = kk - ic * 81;
          int ky = r2 / 9,  kx = r2 - ky * 9;
          long idx = (((long)bimg * CIN + ic) * HIN + (oy * STRIDE + ky)) * WIN
                     + (ox * STRIDE + kx);
          v = IN_F16 ? (float)xh[idx] : xf[idx];
        }
        dst[j] = (half_t)v;
      }
    }
    wait_b_tile();
    __syncthreads();
    v16h bf = lds_frag_b(Bs, wc * 16, lane);
    v16h a0 = lds_frag_a(As, wr * 32, lane);
    v16h a1 = lds_frag_a(As, wr * 32 + 16, lane);
    acc0 = wmma_f16(a0, bf, acc0);
    acc1 = wmma_f16(a1, bf, acc1);
    __syncthreads();
  }

  // epilogue: bias (+relu), scatter to NCHW
  const int nn = n0 + wc * 16 + (lane & 15);
  const float bv = bias[nn];
  const int mb = m0 + wr * 32 + ((lane < 16) ? 0 : 8);
#pragma unroll
  for (int r = 0; r < 8; ++r) {
#pragma unroll
    for (int g = 0; g < 2; ++g) {
      int mm = mb + g * 16 + r;
      float v = (g ? acc1[r] : acc0[r]) + bv;
      if (RELU) v = fmaxf(v, 0.f);
      int bo = mm / OPIX; int po = mm - bo * OPIX;
      long o = ((long)bo * OC + nn) * OPIX + po;
      if (OUT_F16) ((half_t*)yout)[o] = (half_t)v;
      else         ((float*)yout)[o]  = v;
    }
  }
}

// --- dense GEMM:  C[M,N] = act(A[M,K] @ Bw[N,K]^T + bias), Bw in f16 -------
template <int ACT>  // 0 none, 1 relu, 2 sigmoid
__device__ __forceinline__ float activate(float x) {
  if (ACT == 1) return fmaxf(x, 0.f);
  if (ACT == 2) return 1.f / (1.f + __expf(-x));
  return x;
}

template <int ACT>
__global__ __launch_bounds__(256) void dense_wmma_kernel(
    const float* __restrict__ A, const half_t* __restrict__ Bw,
    const float* __restrict__ bias, float* __restrict__ C,
    int M, int N, int K) {
  __shared__ alignas(16) half_t As[TILE_M * LDSK];
  __shared__ alignas(16) half_t Bs[TILE_N * LDSK];

  const int tid  = threadIdx.x;
  const int lane = tid & 31;
  const int wave = tid >> 5;
  const int wr = wave >> 2, wc = wave & 3;
  const int m0 = blockIdx.x * TILE_M;
  const int n0 = blockIdx.y * TILE_N;

  v8f acc0 = {}; v8f acc1 = {};
  const int row = tid >> 2;
  const int kq  = (tid & 3) * 8;

  for (int k0 = 0; k0 < K; k0 += TILE_K) {
    stage_b_tile(Bw, n0, k0, K, N, Bs, row, kq);
    // stage A: vectorized f32->f16 (K % 32 == 0, rows 16B aligned)
    {
      int m = m0 + row;
      union { f32x4 v; half_t h[8]; } o;
      if (m < M) {
        const float* src = A + (long)m * K + k0 + kq;
        f32x4 s0 = *(const f32x4*)src;
        f32x4 s1 = *(const f32x4*)(src + 4);
#pragma unroll
        for (int j = 0; j < 4; ++j) { o.h[j] = (half_t)s0[j]; o.h[4 + j] = (half_t)s1[j]; }
      } else {
#pragma unroll
        for (int j = 0; j < 8; ++j) o.h[j] = (half_t)0.f;
      }
      *(f32x4*)&As[row * LDSK + kq] = o.v;
    }
    wait_b_tile();
    __syncthreads();
    v16h bf = lds_frag_b(Bs, wc * 16, lane);
    v16h a0 = lds_frag_a(As, wr * 32, lane);
    v16h a1 = lds_frag_a(As, wr * 32 + 16, lane);
    acc0 = wmma_f16(a0, bf, acc0);
    acc1 = wmma_f16(a1, bf, acc1);
    __syncthreads();
  }

  const int nn = n0 + wc * 16 + (lane & 15);
  if (nn < N) {
    const float bv = bias[nn];
    const int mb = m0 + wr * 32 + ((lane < 16) ? 0 : 8);
#pragma unroll
    for (int r = 0; r < 8; ++r) {
      int mm0 = mb + r, mm1 = mb + 16 + r;
      if (mm0 < M) C[(long)mm0 * N + nn] = activate<ACT>(acc0[r] + bv);
      if (mm1 < M) C[(long)mm1 * N + nn] = activate<ACT>(acc1[r] + bv);
    }
  }
}

// --- small kernels ---------------------------------------------------------
__global__ void zero_f32_kernel(float* p, long n) {
  long i = (long)blockIdx.x * blockDim.x + threadIdx.x;
  if (i < n) p[i] = 0.f;
}

// f32 (N,K) -> f16 (N,KP) with zero pad for k >= K
__global__ __launch_bounds__(256) void convert_pad_f16_kernel(
    const float* __restrict__ src, half_t* __restrict__ dst, int K, int KP,
    long total) {
  long i = (long)blockIdx.x * 256 + threadIdx.x;
  if (i >= total) return;
  long n = i / KP;
  int k = (int)(i - n * KP);
  dst[i] = (k < K) ? (half_t)src[n * (long)K + k] : (half_t)0.f;
}

// squash over last dim (8) of (B*1152, 8)
__global__ __launch_bounds__(256) void squash_kernel(
    const float* __restrict__ h, float* __restrict__ u, int nvec) {
  int i = blockIdx.x * 256 + threadIdx.x;
  if (i >= nvec) return;
  const float* p = h + (long)i * 8;
  float vals[8]; float sq = 0.f;
#pragma unroll
  for (int d = 0; d < 8; ++d) { vals[d] = p[d]; sq += vals[d] * vals[d]; }
  float scale = sq / ((1.f + sq) * sqrtf(sq + 1e-8f));
  float* q = u + (long)i * 8;
#pragma unroll
  for (int d = 0; d < 8; ++d) q[d] = vals[d] * scale;
}

// u_hat[b,c,n,e] = sum_d u[b,n,d] * W[c,n,d,e]  (stored f16)
__global__ __launch_bounds__(256) void uhat_kernel(
    const float* __restrict__ u, const float* __restrict__ W,
    half_t* __restrict__ uhat) {
  const int tid = threadIdx.x;
  const int n = blockIdx.x * 16 + (tid >> 4);
  const int e = tid & 15;
  const int c = blockIdx.y;
  const int b = blockIdx.z;
  const float* uv = u + ((long)b * 1152 + n) * 8;
  const float* wp = W + (((long)c * 1152 + n) * 8) * 16 + e;
  float s = 0.f;
#pragma unroll
  for (int d = 0; d < 8; ++d) s += uv[d] * wp[d * 16];
  uhat[(((long)b * 10 + c) * 1152 + n) * 16 + e] = (half_t)s;
}

// one dynamic-routing iteration for one (image, class)
__global__ __launch_bounds__(256) void routing_iter_kernel(
    const half_t* __restrict__ uhat, float* __restrict__ logits,
    float* __restrict__ vout, int last) {
  const int NCAP = 1152;
  const int tid = threadIdx.x;
  const int cls = blockIdx.x;
  const int img = blockIdx.y;
  const long base = ((long)img * 10 + cls) * NCAP;

  __shared__ float red[256];
  __shared__ float svec[16];
  __shared__ float vsh[16];
  __shared__ float ssum_sh;

  float lmax = -3.4e38f;
  for (int n = tid; n < NCAP; n += 256) lmax = fmaxf(lmax, logits[base + n]);
  red[tid] = lmax; __syncthreads();
  for (int s = 128; s > 0; s >>= 1) {
    if (tid < s) red[tid] = fmaxf(red[tid], red[tid + s]);
    __syncthreads();
  }
  const float gmax = red[0];
  __syncthreads();

  float lsum = 0.f;
  float sacc[16];
#pragma unroll
  for (int j = 0; j < 16; ++j) sacc[j] = 0.f;
  for (int n = tid; n < NCAP; n += 256) {
    float e = __expf(logits[base + n] - gmax);
    lsum += e;
    const half_t* up = uhat + (base + n) * 16;
#pragma unroll
    for (int j = 0; j < 16; ++j) sacc[j] += e * (float)up[j];
  }
  red[tid] = lsum; __syncthreads();
  for (int s = 128; s > 0; s >>= 1) {
    if (tid < s) red[tid] += red[tid + s];
    __syncthreads();
  }
  if (tid == 0) ssum_sh = red[0];
  if (tid < 16) svec[tid] = 0.f;
  __syncthreads();
#pragma unroll
  for (int j = 0; j < 16; ++j) atomicAdd(&svec[j], sacc[j]);
  __syncthreads();

  if (tid == 0) {
    float inv = 1.f / ssum_sh;
    float sv[16]; float sq = 0.f;
    for (int j = 0; j < 16; ++j) { sv[j] = svec[j] * inv; sq += sv[j] * sv[j]; }
    float scale = sq / ((1.f + sq) * sqrtf(sq + 1e-8f));
    for (int j = 0; j < 16; ++j) {
      float vj = sv[j] * scale;
      vsh[j] = vj;
      if (last) vout[((long)img * 10 + cls) * 16 + j] = vj;
    }
  }
  __syncthreads();

  if (!last) {
    for (int n = tid; n < NCAP; n += 256) {
      const half_t* up = uhat + (base + n) * 16;
      float dot = 0.f;
#pragma unroll
      for (int j = 0; j < 16; ++j) dot += vsh[j] * (float)up[j];
      logits[base + n] += dot;
    }
  }
}

// class probs + argmax mask -> (B,160) decoder input
__global__ __launch_bounds__(160) void mask_kernel(
    const float* __restrict__ v, float* __restrict__ probs_out,
    float* __restrict__ masked) {
  const int b = blockIdx.x;
  const int tid = threadIdx.x;
  __shared__ float norms[10];
  __shared__ int best;
  if (tid < 10) {
    float s = 0.f;
    const float* vp = v + ((long)b * 10 + tid) * 16;
#pragma unroll
    for (int e = 0; e < 16; ++e) s += vp[e] * vp[e];
    norms[tid] = sqrtf(s);
  }
  __syncthreads();
  if (tid == 0) {
    float mx = norms[0]; int bi = 0;
    for (int c = 1; c < 10; ++c) if (norms[c] > mx) { mx = norms[c]; bi = c; }
    float ex[10]; float se = 0.f;
    for (int c = 0; c < 10; ++c) { ex[c] = __expf(norms[c] - mx); se += ex[c]; }
    float inv = 1.f / se;
    for (int c = 0; c < 10; ++c) probs_out[(long)b * 10 + c] = ex[c] * inv;
    best = bi;
  }
  __syncthreads();
  const int c = tid >> 4;
  masked[(long)b * 160 + tid] =
      (c == best) ? v[((long)b * 10 + c) * 16 + (tid & 15)] : 0.f;
}

// ---------------------------------------------------------------------------
extern "C" void kernel_launch(void* const* d_in, const int* in_sizes, int n_in,
                              void* d_out, int out_size, void* d_ws,
                              size_t ws_size, hipStream_t stream) {
  (void)in_sizes; (void)n_in; (void)out_size; (void)ws_size;

  const float* x    = (const float*)d_in[0];
  const float* c1w  = (const float*)d_in[1];
  const float* c1b  = (const float*)d_in[2];
  const float* c2w  = (const float*)d_in[3];
  const float* c2b  = (const float*)d_in[4];
  const float* Wcap = (const float*)d_in[5];
  const float* d1w  = (const float*)d_in[6];
  const float* d1b  = (const float*)d_in[7];
  const float* d2w  = (const float*)d_in[8];
  const float* d2b  = (const float*)d_in[9];
  const float* d3w  = (const float*)d_in[10];
  const float* d3b  = (const float*)d_in[11];
  float* out = (float*)d_out;

  char* ws = (char*)d_ws;
  size_t off = 0;
  auto take = [&](size_t bytes) -> char* {
    char* p = ws + off;
    off = (off + bytes + 255) & ~(size_t)255;
    return p;
  };
  half_t* h1    = (half_t*)take((size_t)512 * 256 * 400 * 2);   // conv1 out f16
  float*  h2    = (float*) take((size_t)512 * 9216 * 4);        // conv2 out
  float*  ucap  = (float*) take((size_t)512 * 9216 * 4);        // squashed u
  half_t* uhat  = (half_t*)take((size_t)512 * 10 * 1152 * 16 * 2);
  float*  logit = (float*) take((size_t)512 * 10 * 1152 * 4);
  float*  vbuf  = (float*) take((size_t)512 * 10 * 16 * 4);
  float*  maskd = (float*) take((size_t)512 * 160 * 4);
  float*  h1d   = (float*) take((size_t)512 * 512 * 4);
  float*  h2d   = (float*) take((size_t)512 * 1024 * 4);
  half_t* c1wh  = (half_t*)take((size_t)256 * 96 * 2);          // f16 weights
  half_t* c2wh  = (half_t*)take((size_t)256 * 20736 * 2);
  half_t* d1wh  = (half_t*)take((size_t)512 * 160 * 2);
  half_t* d2wh  = (half_t*)take((size_t)1024 * 512 * 2);
  half_t* d3wh  = (half_t*)take((size_t)784 * 1024 * 2);

  // weight pre-conversion to f16 (zero-padded K for conv1: 81 -> 96)
  {
    long t;
    t = (long)256 * 96;
    convert_pad_f16_kernel<<<(unsigned)((t + 255) / 256), 256, 0, stream>>>(
        c1w, c1wh, 81, 96, t);
    t = (long)256 * 20736;
    convert_pad_f16_kernel<<<(unsigned)((t + 255) / 256), 256, 0, stream>>>(
        c2w, c2wh, 20736, 20736, t);
    t = (long)512 * 160;
    convert_pad_f16_kernel<<<(unsigned)((t + 255) / 256), 256, 0, stream>>>(
        d1w, d1wh, 160, 160, t);
    t = (long)1024 * 512;
    convert_pad_f16_kernel<<<(unsigned)((t + 255) / 256), 256, 0, stream>>>(
        d2w, d2wh, 512, 512, t);
    t = (long)784 * 1024;
    convert_pad_f16_kernel<<<(unsigned)((t + 255) / 256), 256, 0, stream>>>(
        d3w, d3wh, 1024, 1024, t);
  }

  // conv1: (512,1,28,28) -> (512,256,20,20), relu, f16 out.  M = 512*400
  conv_igemm_wmma_kernel<1, 28, 28, 20, 20, 1, 204800, true, false, true>
      <<<dim3(204800 / 64, 4), 256, 0, stream>>>(x, c1wh, c1b, h1);

  // conv2: (512,256,20,20) -> (512,256,6,6), no act, f32 out.  M = 512*36
  conv_igemm_wmma_kernel<256, 20, 20, 6, 6, 2, 18432, false, true, false>
      <<<dim3(18432 / 64, 4), 256, 0, stream>>>(h1, c2wh, c2b, h2);

  // squash -> u  (512*1152 vectors of 8)
  squash_kernel<<<(589824 + 255) / 256, 256, 0, stream>>>(h2, ucap, 589824);

  // u_hat = einsum('bnd,cnde->bcne')
  uhat_kernel<<<dim3(72, 10, 512), 256, 0, stream>>>(ucap, Wcap, uhat);

  // routing logits = 0, then 3 routing iterations
  {
    long nl = (long)512 * 10 * 1152;
    zero_f32_kernel<<<(unsigned)((nl + 255) / 256), 256, 0, stream>>>(logit, nl);
  }
  for (int it = 0; it < 3; ++it)
    routing_iter_kernel<<<dim3(10, 512), 256, 0, stream>>>(uhat, logit, vbuf,
                                                           it == 2 ? 1 : 0);

  // class_probs (to out[0:5120]) + masked decoder input
  mask_kernel<<<512, 160, 0, stream>>>(vbuf, out, maskd);

  // decoder: 160->512 relu, 512->1024 relu, 1024->784 sigmoid (to out+5120)
  dense_wmma_kernel<1><<<dim3(8, 8),  256, 0, stream>>>(maskd, d1wh, d1b, h1d,
                                                        512, 512, 160);
  dense_wmma_kernel<1><<<dim3(8, 16), 256, 0, stream>>>(h1d, d2wh, d2b, h2d,
                                                        512, 1024, 512);
  dense_wmma_kernel<2><<<dim3(8, 13), 256, 0, stream>>>(h2d, d3wh, d3b,
                                                        out + 5120,
                                                        512, 784, 1024);
}
